// Agent_19963007992380
// MI455X (gfx1250) — matrix operand order repaired and checked by reading, not compiled, not run on hardware
//
#include <hip/hip_runtime.h>
#include <hip/hip_bf16.h>

typedef __attribute__((ext_vector_type(16))) __bf16 v16bf;
typedef __attribute__((ext_vector_type(8)))  float  v8f;

#define S_LEN 512
#define BATCH 256
#define OBSD  180
#define ACTD  64
#define HD    256
#define G4    1024
#define NWG   16

// ---------------------------------------------------------------------------
// Fragment loaders (WMMA 16x16x32 bf16; contiguous k order used by this file for both A and B)
// A 16x32: lane L -> M=L%16; dwords 0..7: K = 16*(L/16)+{0..15}
// B 32x16: lane L -> N=L%16; dwords 0..7: K = 16*(L/16)+{0..15}
// ---------------------------------------------------------------------------
__device__ __forceinline__ v16bf load_fragA(const __bf16* lds, int row, int ldk,
                                            int kb, int half) {
  v16bf f;
  unsigned* fp = reinterpret_cast<unsigned*>(&f);
  const unsigned* lo = reinterpret_cast<const unsigned*>(lds + row * ldk + kb + 16 * half);
  const unsigned* hi = lo + 4;
  fp[0] = lo[0]; fp[1] = lo[1]; fp[2] = lo[2]; fp[3] = lo[3];
  fp[4] = hi[0]; fp[5] = hi[1]; fp[6] = hi[2]; fp[7] = hi[3];
  return f;
}

__device__ __forceinline__ v16bf load_fragB(const __bf16* lds, int row, int ldk,
                                            int kb, int half) {
  v16bf f;
  unsigned* fp = reinterpret_cast<unsigned*>(&f);
  const unsigned* p = reinterpret_cast<const unsigned*>(lds + row * ldk + kb + 16 * half);
#pragma unroll
  for (int v = 0; v < 8; ++v) fp[v] = p[v];
  return f;
}

__device__ __forceinline__ v8f wmma_bf16(v16bf a, v16bf b, v8f c) {
  return __builtin_amdgcn_wmma_f32_16x16x32_bf16(false, a, false, b, (short)0, c,
                                                 false, false);
}

__device__ __forceinline__ float sigm(float x) { return 1.0f / (1.0f + __expf(-x)); }

// ---------------------------------------------------------------------------
// Tile staging helpers (vectorized: all K here are multiples of 4)
// ---------------------------------------------------------------------------
__device__ __forceinline__ void stage_A(const float* __restrict__ A, __bf16* As,
                                        int bm, int k0, int M, int K, int tid) {
#pragma unroll
  for (int i = 0; i < 4; ++i) {
    int g = tid + 256 * i;            // 1024 float4 groups (128x32)
    int r = g >> 3, c4 = (g & 7) * 4;
    int gm = bm + r, gk = k0 + c4;
    float4 v = {0.f, 0.f, 0.f, 0.f};
    if (gm < M) {
      if (gk + 3 < K) {
        v = *(const float4*)(A + (long long)gm * K + gk);
      } else {
        const float* row = A + (long long)gm * K;
        if (gk + 0 < K) v.x = row[gk + 0];
        if (gk + 1 < K) v.y = row[gk + 1];
        if (gk + 2 < K) v.z = row[gk + 2];
        if (gk + 3 < K) v.w = row[gk + 3];
      }
    }
    __bf16* d = As + r * 32 + c4;
    d[0] = (__bf16)v.x; d[1] = (__bf16)v.y; d[2] = (__bf16)v.z; d[3] = (__bf16)v.w;
  }
}

__device__ __forceinline__ void stage_A(const __bf16* __restrict__ A, __bf16* As,
                                        int bm, int k0, int M, int K, int tid) {
#pragma unroll
  for (int i = 0; i < 2; ++i) {
    int g = tid + 256 * i;            // 512 uint4 groups (8 bf16 each)
    int r = g >> 2, c8 = (g & 3) * 8;
    int gm = bm + r, gk = k0 + c8;
    uint4 v = {0u, 0u, 0u, 0u};
    if (gm < M && gk + 7 < K) v = *(const uint4*)(A + (long long)gm * K + gk);
    *(uint4*)(As + r * 32 + c8) = v;
  }
}

__device__ __forceinline__ void stage_B(const float* __restrict__ W, __bf16* Bs,
                                        int bn, int k0, int N, int K, int tid) {
#pragma unroll
  for (int i = 0; i < 4; ++i) {
    int g = tid + 256 * i;            // 1024 float4 groups (128x32)
    int r = g >> 3, c4 = (g & 7) * 4;
    int gn = bn + r, gk = k0 + c4;
    float4 v = {0.f, 0.f, 0.f, 0.f};
    if (gn < N) {
      if (gk + 3 < K) {
        v = *(const float4*)(W + (long long)gn * K + gk);
      } else {
        const float* row = W + (long long)gn * K;
        if (gk + 0 < K) v.x = row[gk + 0];
        if (gk + 1 < K) v.y = row[gk + 1];
        if (gk + 2 < K) v.z = row[gk + 2];
        if (gk + 3 < K) v.w = row[gk + 3];
      }
    }
    __bf16* d = Bs + r * 32 + c4;
    d[0] = (__bf16)v.x; d[1] = (__bf16)v.y; d[2] = (__bf16)v.z; d[3] = (__bf16)v.w;
  }
}

// ---------------------------------------------------------------------------
// Generic tiled GEMM: C[M,N] = act(A[M,K] @ W[N,K]^T + bias (+bias2))
// Block tile 128x128, 8 waves (2Mx4N); wave tile 64x32 = 4x2 WMMA tiles
// (8 WMMA per K-chunk from 6 fragment loads).
// ---------------------------------------------------------------------------
template <typename AT, typename OT, int ACTMODE>
__global__ __launch_bounds__(256) void gemm_bias_act(
    const AT* __restrict__ A, const float* __restrict__ W,
    const float* __restrict__ bias, const float* __restrict__ bias2,
    OT* __restrict__ C, int M, int N, int K) {
  __shared__ __bf16 As[128 * 32];
  __shared__ __bf16 Bs[128 * 32];

  const int tid  = threadIdx.x;
  const int lane = tid & 31;
  const int wave = tid >> 5;
  const int half = lane >> 4;
  const int lm   = lane & 15;
  const int bm   = blockIdx.x * 128;
  const int bn   = blockIdx.y * 128;
  const int wm   = (wave >> 2) * 64;
  const int wn   = (wave & 3) * 32;

  v8f acc[4][2];
#pragma unroll
  for (int i = 0; i < 4; ++i)
#pragma unroll
    for (int j = 0; j < 2; ++j) acc[i][j] = {};

  for (int k0 = 0; k0 < K; k0 += 32) {
    stage_A(A, As, bm, k0, M, K, tid);
    stage_B(W, Bs, bn, k0, N, K, tid);
    __syncthreads();

    v16bf b0 = load_fragB(Bs, wn + lm, 32, 0, half);
    v16bf b1 = load_fragB(Bs, wn + 16 + lm, 32, 0, half);
#pragma unroll
    for (int tm = 0; tm < 4; ++tm) {
      v16bf a = load_fragA(As, wm + tm * 16 + lm, 32, 0, half);
      acc[tm][0] = wmma_bf16(a, b0, acc[tm][0]);
      acc[tm][1] = wmma_bf16(a, b1, acc[tm][1]);
    }
    __syncthreads();
  }

#pragma unroll
  for (int tm = 0; tm < 4; ++tm)
#pragma unroll
    for (int tn = 0; tn < 2; ++tn) {
      int n = bn + wn + tn * 16 + lm;
      float bs = 0.0f;
      if (n < N) {
        bs = bias ? bias[n] : 0.0f;
        if (bias2) bs += bias2[n];
      }
#pragma unroll
      for (int r = 0; r < 8; ++r) {
        int m = bm + wm + tm * 16 + r + 8 * half;
        if (m < M && n < N) {
          float v = acc[tm][tn][r] + bs;
          if (ACTMODE == 1) v = tanhf(v);
          C[(long long)m * N + n] = (OT)v;
        }
      }
    }
}

// ---------------------------------------------------------------------------
// Device-scope barrier across the persistent LSTM workgroups.
// Release: fence (global_wb) before arrive; Acquire: fence (global_inv) after.
// ---------------------------------------------------------------------------
__device__ __forceinline__ void grid_barrier(unsigned* cnt, unsigned* gen,
                                             unsigned nwg, unsigned genval) {
  __threadfence();
  __syncthreads();
  if (threadIdx.x == 0) {
    unsigned prev = atomicAdd(cnt, 1u);
    if (prev == nwg - 1u) {
      atomicExch(cnt, 0u);
      atomicExch(gen, genval);
    } else {
      while (atomicAdd(gen, 0u) < genval) __builtin_amdgcn_s_sleep(2);
    }
  }
  __syncthreads();
  __threadfence();  // acquire: subsequent hbuf reads must miss stale L0/WGP$
}

// ---------------------------------------------------------------------------
// Persistent LSTM scan: NWG=16 workgroups, each owns a 16-wide hidden slice
// (all 4 gates for those units -> elementwise update is WG-local).
// Per step: gates = xg[t] + h @ whh^T  (M=256 batch, N=64 slice cols, K=256).
// LDS: full h (256x256 bf16, 128KB) + whh slice (64x256 bf16, 32KB)
//      + c slice (256x16 f32, 16KB) = 176KB (CDNA5 320KB WGP LDS).
// xg lines for step t+1 are prefetched (global_prefetch_b8) during step t.
// ---------------------------------------------------------------------------
__global__ __launch_bounds__(256) void lstm_scan(
    const float* __restrict__ xg,       // [S,B,1024] gate preactivations
    const __bf16* __restrict__ whh_bf,  // [1024,256]
    const float* __restrict__ h0,       // [B,256]
    const float* __restrict__ c0,       // [B,256]
    __bf16* __restrict__ y,             // [S,B,256] out (bf16 for next GEMM)
    __bf16* __restrict__ hbuf,          // [B,256] cross-WG h exchange
    float* __restrict__ hn_out,         // [B,256] final h (f32)
    float* __restrict__ cn_out,         // [B,256] final c (f32)
    unsigned* __restrict__ bar_cnt, unsigned* __restrict__ bar_gen,
    unsigned gen_base) {
  extern __shared__ char smem[];
  __bf16* hA  = (__bf16*)smem;                      // 256*256 bf16
  __bf16* whS = (__bf16*)(smem + 131072);           // 64*256 bf16
  float*  cS  = (float*)(smem + 131072 + 32768);    // 256*16 f32

  const int tid  = threadIdx.x;
  const int lane = tid & 31;
  const int wave = tid >> 5;
  const int half = lane >> 4;
  const int lm   = lane & 15;
  const int j0   = blockIdx.x * 16;  // hidden slice
  const int mb   = wave * 32;        // batch rows of this wave

  // Resident whh slice: LDS col c -> gate row (c/16)*256 + j0 + c%16
  {
    const uint4* src = (const uint4*)whh_bf;  // 32 uint4 per 256-col row
    uint4* dst = (uint4*)whS;
#pragma unroll
    for (int i = 0; i < 8; ++i) {
      int g = tid + 256 * i;  // 2048 uint4
      int c = g >> 5, kk = g & 31;
      int gidx = (c >> 4) * 256 + j0 + (c & 15);
      dst[c * 32 + kk] = src[gidx * 32 + kk];
    }
  }
  // Initial h (f32 -> bf16 LDS) and c slice
  for (int i = 0; i < 64; ++i) {
    int g = tid + 256 * i;  // 16384 float4
    float4 v = ((const float4*)h0)[g];
    __bf16* d = hA + g * 4;
    d[0] = (__bf16)v.x; d[1] = (__bf16)v.y; d[2] = (__bf16)v.z; d[3] = (__bf16)v.w;
  }
#pragma unroll
  for (int i = 0; i < 16; ++i) {
    int lin = tid + 256 * i;  // b = lin/16, jl = lin%16
    cS[lin] = c0[(lin >> 4) * 256 + j0 + (lin & 15)];
  }
  __syncthreads();

  unsigned gen = gen_base;
  for (int t = 0; t < S_LEN; ++t) {
    const float* xgt = xg + (long long)t * (BATCH * G4);

    // Seed gate accumulators from precomputed xg[t] (WMMA C-operand layout)
    v8f acc[2][4];
#pragma unroll
    for (int tm = 0; tm < 2; ++tm)
#pragma unroll
      for (int g = 0; g < 4; ++g) {
        int n = g * 256 + j0 + lm;
#pragma unroll
        for (int r = 0; r < 8; ++r) {
          int b = mb + tm * 16 + r + 8 * half;
          acc[tm][g][r] = xgt[(long long)b * G4 + n];
        }
      }

    // Prefetch next step's xg slice (fire-and-forget, no counter, no VGPRs):
    // lanes 0..31 of each prefetch cover both 64B lines (half=0/1 rows).
    if (t + 1 < S_LEN) {
      const float* xgn = xgt + BATCH * G4;
#pragma unroll
      for (int tm = 0; tm < 2; ++tm)
#pragma unroll
        for (int g = 0; g < 4; ++g) {
          int n = g * 256 + j0 + lm;
#pragma unroll
          for (int r = 0; r < 8; ++r) {
            int b = mb + tm * 16 + r + 8 * half;
            __builtin_prefetch(xgn + (long long)b * G4 + n, 0, 3);
          }
        }
    }

    // gates += h @ whh_slice^T  (K = 256, 8 chunks, 8 WMMA each)
#pragma unroll
    for (int kc = 0; kc < 8; ++kc) {
      int kb = kc * 32;
      v16bf bfr[4];
#pragma unroll
      for (int g = 0; g < 4; ++g) bfr[g] = load_fragB(whS, g * 16 + lm, 256, kb, half);
#pragma unroll
      for (int tm = 0; tm < 2; ++tm) {
        v16bf af = load_fragA(hA, mb + tm * 16 + lm, 256, kb, half);
#pragma unroll
        for (int g = 0; g < 4; ++g) acc[tm][g] = wmma_bf16(af, bfr[g], acc[tm][g]);
      }
    }

    // Elementwise LSTM cell update (i,f,g,o all live in this wave's accs)
#pragma unroll
    for (int tm = 0; tm < 2; ++tm)
#pragma unroll
      for (int r = 0; r < 8; ++r) {
        int b = mb + tm * 16 + r + 8 * half;
        float iv = sigm(acc[tm][0][r]);
        float fv = sigm(acc[tm][1][r]);
        float gv = tanhf(acc[tm][2][r]);
        float ov = sigm(acc[tm][3][r]);
        float cn = fv * cS[b * 16 + lm] + iv * gv;
        cS[b * 16 + lm] = cn;
        float hv = ov * tanhf(cn);
        __bf16 hb = (__bf16)hv;
        y[((long long)t * BATCH + b) * HD + j0 + lm] = hb;
        hbuf[b * HD + j0 + lm] = hb;
        if (t == S_LEN - 1) {
          hn_out[b * HD + j0 + lm] = hv;
          cn_out[b * HD + j0 + lm] = cn;
        }
      }

    if (t < S_LEN - 1) {
      grid_barrier(bar_cnt, bar_gen, NWG, ++gen);
      // Pull the full updated h into LDS for the next step (16B vectors)
      const uint4* src = (const uint4*)hbuf;
      uint4* dst = (uint4*)hA;
      for (int i = 0; i < 32; ++i) dst[tid + 256 * i] = src[tid + 256 * i];
      __syncthreads();
    }
  }
}

// ---------------------------------------------------------------------------
// Value head: values[m] = dot(y1[m,:], wv) + bv  (one row per wave)
// ---------------------------------------------------------------------------
__global__ __launch_bounds__(256) void values_kernel(
    const __bf16* __restrict__ y1, const float* __restrict__ wv,
    const float* __restrict__ bv, float* __restrict__ out) {
  int row  = blockIdx.x * 8 + (threadIdx.x >> 5);
  int lane = threadIdx.x & 31;
  const __bf16* yr = y1 + (long long)row * HD + lane * 8;
  const float* wr = wv + lane * 8;
  float s = 0.0f;
#pragma unroll
  for (int i = 0; i < 8; ++i) s += (float)yr[i] * wr[i];
#pragma unroll
  for (int off = 16; off; off >>= 1) s += __shfl_xor(s, off, 32);
  if (lane == 0) out[row] = s + bv[0];
}

__global__ void cvt_bf16_kernel(const float* __restrict__ src,
                                __bf16* __restrict__ dst, int n) {
  int i = blockIdx.x * 256 + threadIdx.x;
  if (i < n) dst[i] = (__bf16)src[i];
}

// ---------------------------------------------------------------------------
extern "C" void kernel_launch(void* const* d_in, const int* in_sizes, int n_in,
                              void* d_out, int out_size, void* d_ws,
                              size_t ws_size, hipStream_t stream) {
  (void)in_sizes; (void)n_in; (void)out_size; (void)ws_size;

  const float* inputs = (const float*)d_in[0];
  const float* w1   = (const float*)d_in[1];
  const float* b1   = (const float*)d_in[2];
  const float* w2   = (const float*)d_in[3];
  const float* b2   = (const float*)d_in[4];
  const float* wih0 = (const float*)d_in[5];
  const float* whh0 = (const float*)d_in[6];
  const float* bih0 = (const float*)d_in[7];
  const float* bhh0 = (const float*)d_in[8];
  const float* wih1 = (const float*)d_in[9];
  const float* whh1 = (const float*)d_in[10];
  const float* bih1 = (const float*)d_in[11];
  const float* bhh1 = (const float*)d_in[12];
  const float* wp   = (const float*)d_in[13];
  const float* bp   = (const float*)d_in[14];
  const float* wv   = (const float*)d_in[15];
  const float* bv   = (const float*)d_in[16];
  const float* h0   = (const float*)d_in[17];
  const float* c0   = (const float*)d_in[18];

  const int M = S_LEN * BATCH;  // 131072

  // Workspace layout
  char* ws = (char*)d_ws;
  __bf16* t0     = (__bf16*)ws;                          // M*H bf16 (67MB)
  __bf16* t1     = (__bf16*)(ws + (size_t)67108864);     // M*H bf16 (67MB)
  float*  xg     = (float*)(ws + (size_t)134217728);     // M*4H f32 (536MB)
  __bf16* whh0b  = (__bf16*)(ws + (size_t)671088640);    // 1024*256 bf16
  __bf16* whh1b  = (__bf16*)(ws + (size_t)671612928);
  __bf16* hbuf   = (__bf16*)(ws + (size_t)672137216);    // B*H bf16
  unsigned* bar  = (unsigned*)(ws + (size_t)672268288);  // barrier cnt/gen

  float* out_logits = (float*)d_out;
  float* out_values = out_logits + (long long)M * ACTD;
  float* out_hn     = out_values + (long long)M;
  float* out_cn     = out_hn + 2 * BATCH * HD;

  hipMemsetAsync(bar, 0, 2 * sizeof(unsigned), stream);
  cvt_bf16_kernel<<<1024, 256, 0, stream>>>(whh0, whh0b, G4 * HD);
  cvt_bf16_kernel<<<1024, 256, 0, stream>>>(whh1, whh1b, G4 * HD);

  dim3 blk(256);
  // MLP: feats = tanh(tanh(inputs@w1.T+b1)@w2.T+b2)
  gemm_bias_act<float, __bf16, 1><<<dim3(M / 128, 2), blk, 0, stream>>>(
      inputs, w1, b1, nullptr, t0, M, HD, OBSD);
  gemm_bias_act<__bf16, __bf16, 1><<<dim3(M / 128, 2), blk, 0, stream>>>(
      t0, w2, b2, nullptr, t1, M, HD, HD);

  size_t smem = 131072 + 32768 + 16384;  // 176KB
  // Layer 0
  gemm_bias_act<__bf16, float, 0><<<dim3(M / 128, 8), blk, 0, stream>>>(
      t1, wih0, bih0, bhh0, xg, M, G4, HD);
  lstm_scan<<<NWG, 256, smem, stream>>>(xg, whh0b, h0, c0, t1, hbuf, out_hn,
                                        out_cn, bar, bar + 1, 0u);
  // Layer 1
  gemm_bias_act<__bf16, float, 0><<<dim3(M / 128, 8), blk, 0, stream>>>(
      t1, wih1, bih1, bhh1, xg, M, G4, HD);
  lstm_scan<<<NWG, 256, smem, stream>>>(xg, whh1b, h0 + BATCH * HD,
                                        c0 + BATCH * HD, t0, hbuf,
                                        out_hn + BATCH * HD,
                                        out_cn + BATCH * HD, bar, bar + 1, 511u);
  // Heads
  gemm_bias_act<__bf16, float, 0><<<dim3(M / 128, 1), blk, 0, stream>>>(
      t0, wp, bp, nullptr, out_logits, M, ACTD, HD);
  values_kernel<<<M / 8, 256, 0, stream>>>(t0, wv, bv, out_values);
}
